// GNN_env_43800076485423
// MI455X (gfx1250) — matrix operand order repaired and checked by reading, not compiled, not run on hardware
//
#include <hip/hip_runtime.h>

#define HIDN 64
#define NFD 20
#define EFD 20
#define NENV 16
#define NNODES 5000
#define NEDGES 50000
#define E2 (2 * NEDGES)
#define NTILES (E2 / 16)                 // 6250
#define TOTAL_TILES (NENV * NTILES)      // 100000
#define LS 72                            // LDS row stride in halves (bank-conflict-free, 16B aligned)

typedef __bf16 v16bf __attribute__((ext_vector_type(16)));
typedef __bf16 v8bf  __attribute__((ext_vector_type(8)));
typedef __bf16 v4bf  __attribute__((ext_vector_type(4)));
typedef float  v8f   __attribute__((ext_vector_type(8)));
typedef float  v4f   __attribute__((ext_vector_type(4)));

// ---- fast transcendentals (branch-free; v_tanh_f32 if available) ----
__device__ __forceinline__ float fast_tanh(float x) {
#if defined(__gfx1250__) && __has_builtin(__builtin_amdgcn_tanhf)
  return __builtin_amdgcn_tanhf(x);
#else
  float ax = __builtin_fabsf(x);
  float t = __builtin_amdgcn_exp2f(ax * -2.885390082f);   // exp(-2|x|)
  float r = (1.0f - t) * __builtin_amdgcn_rcpf(1.0f + t);
  return __builtin_copysignf(r, x);
#endif
}

__device__ __forceinline__ float fast_sigmoid(float x) {
  return __builtin_amdgcn_rcpf(1.0f + __builtin_amdgcn_exp2f(x * -1.442695041f));
}

// A-matrix fragment (16x32 bf16, MxK): lane m=lane&15, element e<8 -> K=kbase+e,
// element e>=8 -> K=kbase+16+(e-8), kbase=(lane>>4)*8 + kt*32.  Two 16B LDS loads.
__device__ __forceinline__ v16bf frag_a(const __bf16* buf, int lane, int kt) {
  int m = lane & 15;
  int k0 = kt * 32 + ((lane >> 4) << 3);
  const __bf16* p = buf + m * LS + k0;
  v8bf lo = *(const v8bf*)p;
  v8bf hi = *(const v8bf*)(p + 16);
  return __builtin_shufflevector(lo, hi, 0, 1, 2, 3, 4, 5, 6, 7, 8, 9, 10, 11, 12, 13, 14, 15);
}

// B-matrix fragment (32x16 bf16, KxN): lane n=lane&15 (+16*nt), element e<8 -> K=kbase+e,
// element e>=8 -> K=kbase+16+(e-8), kbase=(lane>>4)*8 + kt*32.  Weights stored [n][k] row-major (stride LS).
__device__ __forceinline__ v16bf frag_b(const __bf16* w, int lane, int kt, int nt) {
  int n = nt * 16 + (lane & 15);
  int kc = kt * 32 + ((lane >> 4) << 3);
  const __bf16* p = w + n * LS + kc;
  v8bf lo = *(const v8bf*)p;
  v8bf hi = *(const v8bf*)(p + 16);
  return __builtin_shufflevector(lo, hi, 0, 1, 2, 3, 4, 5, 6, 7, 8, 9, 10, 11, 12, 13, 14, 15);
}

__global__ void gnn_zero_kernel(float* __restrict__ p, int n) {
  int i = blockIdx.x * blockDim.x + threadIdx.x;
  if (i < n) p[i] = 0.0f;
}

__global__ __launch_bounds__(256)
void gnn_msg_kernel(const float* __restrict__ nf, const int* __restrict__ edges,
                    const float* __restrict__ W1, const float* __restrict__ b1,
                    const float* __restrict__ W2, const float* __restrict__ b2,
                    const float* __restrict__ W3, const float* __restrict__ b3,
                    float* __restrict__ store) {
  __shared__ __bf16 w1s[64 * LS];
  __shared__ __bf16 w2s[64 * LS];
  __shared__ __bf16 w3s[32 * LS];
  __shared__ float b1s[64], b2s[64], b3s[32];
  __shared__ __bf16 xb[8][16 * LS];
  __shared__ __bf16 hb[8][16 * LS];
  __shared__ int sd[8][16][2];

  int tid = threadIdx.x;
  // Stage weights: f32 -> bf16, K padded 40->64 (W1) and N padded 20->32 (W3).
  for (int i = tid; i < 64 * 64; i += 256) {
    int n = i >> 6, k = i & 63;
    w1s[n * LS + k] = (k < 2 * NFD) ? (__bf16)W1[n * (2 * NFD) + k] : (__bf16)0.0f;
    w2s[n * LS + k] = (__bf16)W2[n * HIDN + k];
  }
  for (int i = tid; i < 32 * 64; i += 256) {
    int n = i >> 6, k = i & 63;
    w3s[n * LS + k] = (n < EFD) ? (__bf16)W3[n * HIDN + k] : (__bf16)0.0f;
  }
  if (tid < 64) { b1s[tid] = b1[tid]; b2s[tid] = b2[tid]; }
  if (tid < 32) b3s[tid] = (tid < EFD) ? b3[tid] : 0.0f;
  __syncthreads();

  int wave = tid >> 5, lane = tid & 31;
  int mb = (lane >> 4) << 3;             // C/D row base: M = mb + v
  const int* e0 = edges;
  const int* e1 = edges + NEDGES;
  __bf16* X = xb[wave];
  __bf16* H = hb[wave];

  // Hoist all loop-invariant B (weight) fragments into registers: 20 x v16bf.
  v16bf b1f[4][2], b2f[4][2], b3f[2][2];
  #pragma unroll
  for (int nt = 0; nt < 4; ++nt) {
    b1f[nt][0] = frag_b(w1s, lane, 0, nt);
    b1f[nt][1] = frag_b(w1s, lane, 1, nt);
    b2f[nt][0] = frag_b(w2s, lane, 0, nt);
    b2f[nt][1] = frag_b(w2s, lane, 1, nt);
  }
  #pragma unroll
  for (int nt = 0; nt < 2; ++nt) {
    b3f[nt][0] = frag_b(w3s, lane, 0, nt);
    b3f[nt][1] = frag_b(w3s, lane, 1, nt);
  }
  float bb1 = b1s[lane & 15];            // per-lane bias, indexed by nt*16 later
  // (biases depend on absolute n = nt*16 + lane&15; keep LDS reads, they're tiny)

  #pragma unroll 1
  for (int t = blockIdx.x * 8 + wave; t < TOTAL_TILES; t += gridDim.x * 8) {
    int env = t / NTILES;
    int ebase = (t % NTILES) << 4;
    (void)bb1;

    if (lane < 16) {
      int e = ebase + lane;
      int s, d;
      if (e < NEDGES) { s = e0[e]; d = e1[e]; }
      else            { s = e1[e - NEDGES]; d = e0[e - NEDGES]; }
      sd[wave][lane][0] = s;
      sd[wave][lane][1] = d;
    }
    __asm__ volatile("s_wait_dscnt 0" ::: "memory");

    // Gather node features for 16 edges -> X tile (16 x 64 bf16, cols 40..63 = 0).
    // float4 global loads -> 2x v_cvt_pk_bf16_f32 -> b64 LDS store (4 cols/iter).
    const float* nfe = nf + (size_t)env * NNODES * NFD;
    #pragma unroll
    for (int i = lane; i < 160; i += 32) {       // 5 iterations
      int row = i / 10, c = i % 10;
      const float* q;
      int xcol;
      if (c < 5) { q = nfe + (size_t)sd[wave][row][0] * NFD + 4 * c;       xcol = 4 * c; }
      else       { q = nfe + (size_t)sd[wave][row][1] * NFD + 4 * (c - 5); xcol = 20 + 4 * (c - 5); }
      v4f f = *(const v4f*)q;
      v4bf pk = {(__bf16)f[0], (__bf16)f[1], (__bf16)f[2], (__bf16)f[3]};
      *(v4bf*)(X + row * LS + xcol) = pk;
    }
    // Zero the K-pad columns 40..63 (L2 writeback dirties them each tile).
    v8bf z8 = {};
    #pragma unroll
    for (int i = lane; i < 48; i += 32) {        // <=2 iterations
      int row = i / 3, c = i % 3;
      *(v8bf*)(X + row * LS + 40 + 8 * c) = z8;
    }
    __asm__ volatile("s_wait_dscnt 0" ::: "memory");

    // Layer 1: H = tanh(X @ W1^T + b1)   (16x64, K=64)
    {
      v16bf a0 = frag_a(X, lane, 0);
      v16bf a1 = frag_a(X, lane, 1);
      #pragma unroll
      for (int nt = 0; nt < 4; ++nt) {
        v8f c = {};
        c = __builtin_amdgcn_wmma_f32_16x16x32_bf16(false, a0, false, b1f[nt][0], (short)0, c, false, false);
        c = __builtin_amdgcn_wmma_f32_16x16x32_bf16(false, a1, false, b1f[nt][1], (short)0, c, false, false);
        int n = nt * 16 + (lane & 15);
        float bb = b1s[n];
        #pragma unroll
        for (int v = 0; v < 8; ++v)
          H[(mb + v) * LS + n] = (__bf16)fast_tanh(c[v] + bb);
      }
    }
    __asm__ volatile("s_wait_dscnt 0" ::: "memory");

    // Layer 2: X = tanh(H @ W2^T + b2)   (16x64, K=64)  (reuse X tile as H2)
    {
      v16bf a0 = frag_a(H, lane, 0);
      v16bf a1 = frag_a(H, lane, 1);
      #pragma unroll
      for (int nt = 0; nt < 4; ++nt) {
        v8f c = {};
        c = __builtin_amdgcn_wmma_f32_16x16x32_bf16(false, a0, false, b2f[nt][0], (short)0, c, false, false);
        c = __builtin_amdgcn_wmma_f32_16x16x32_bf16(false, a1, false, b2f[nt][1], (short)0, c, false, false);
        int n = nt * 16 + (lane & 15);
        float bb = b2s[n];
        #pragma unroll
        for (int v = 0; v < 8; ++v)
          X[(mb + v) * LS + n] = (__bf16)fast_tanh(c[v] + bb);
      }
    }
    __asm__ volatile("s_wait_dscnt 0" ::: "memory");

    // Layer 3: msg = X @ W3^T + b3 (N=20 padded to 32), scatter-add to src nodes
    {
      v16bf a0 = frag_a(X, lane, 0);
      v16bf a1 = frag_a(X, lane, 1);
      float* se = store + (size_t)env * NNODES * EFD;
      float* rowp[8];
      #pragma unroll
      for (int v = 0; v < 8; ++v)
        rowp[v] = se + (size_t)sd[wave][mb + v][0] * EFD;
      #pragma unroll
      for (int nt = 0; nt < 2; ++nt) {
        v8f c = {};
        c = __builtin_amdgcn_wmma_f32_16x16x32_bf16(false, a0, false, b3f[nt][0], (short)0, c, false, false);
        c = __builtin_amdgcn_wmma_f32_16x16x32_bf16(false, a1, false, b3f[nt][1], (short)0, c, false, false);
        int n = nt * 16 + (lane & 15);
        if (n < EFD) {
          float bb = b3s[n];
          #pragma unroll
          for (int v = 0; v < 8; ++v)
            atomicAdd(rowp[v] + n, c[v] + bb);
        }
      }
    }
    __asm__ volatile("s_wait_dscnt 0" ::: "memory");
  }
}

__global__ __launch_bounds__(256)
void gnn_gru_kernel(const float* __restrict__ store, const float* __restrict__ nf,
                    const float* __restrict__ Wih, const float* __restrict__ Whh,
                    const float* __restrict__ bih, const float* __restrict__ bhh,
                    float* __restrict__ out) {
  __shared__ float wih_s[60 * 20], whh_s[60 * 20], bih_s[60], bhh_s[60];
  int tid = threadIdx.x;
  for (int i = tid; i < 1200; i += 256) { wih_s[i] = Wih[i]; whh_s[i] = Whh[i]; }
  if (tid < 60) { bih_s[tid] = bih[tid]; bhh_s[tid] = bhh[tid]; }
  __syncthreads();

  int gid = blockIdx.x * 256 + tid;
  if (gid >= NENV * NNODES) return;
  float mv[NFD], hv[NFD];
  const float* mrow = store + (size_t)gid * EFD;
  const float* hrow = nf + (size_t)gid * NFD;
  #pragma unroll
  for (int k = 0; k < NFD; ++k) { mv[k] = mrow[k]; hv[k] = hrow[k]; }

  float* orow = out + (size_t)gid * NFD;
  for (int j = 0; j < NFD; ++j) {
    float ir = bih_s[j], iz = bih_s[j + 20], in = bih_s[j + 40];
    float hr = bhh_s[j], hz = bhh_s[j + 20], hn = bhh_s[j + 40];
    #pragma unroll
    for (int k = 0; k < NFD; ++k) {
      ir += wih_s[j * 20 + k] * mv[k];
      iz += wih_s[(j + 20) * 20 + k] * mv[k];
      in += wih_s[(j + 40) * 20 + k] * mv[k];
      hr += whh_s[j * 20 + k] * hv[k];
      hz += whh_s[(j + 20) * 20 + k] * hv[k];
      hn += whh_s[(j + 40) * 20 + k] * hv[k];
    }
    float r = fast_sigmoid(ir + hr);
    float z = fast_sigmoid(iz + hz);
    float n = fast_tanh(in + r * hn);
    orow[j] = (1.0f - z) * n + z * hv[j];
  }
}

extern "C" void kernel_launch(void* const* d_in, const int* in_sizes, int n_in,
                              void* d_out, int out_size, void* d_ws, size_t ws_size,
                              hipStream_t stream) {
  const float* nf_in = (const float*)d_in[0];
  const int*   edges = (const int*)d_in[1];
  const float* W1 = (const float*)d_in[2];
  const float* b1 = (const float*)d_in[3];
  const float* W2 = (const float*)d_in[4];
  const float* b2 = (const float*)d_in[5];
  const float* W3 = (const float*)d_in[6];
  const float* b3 = (const float*)d_in[7];
  const float* Wih = (const float*)d_in[8];
  const float* Whh = (const float*)d_in[9];
  const float* bih = (const float*)d_in[10];
  const float* bhh = (const float*)d_in[11];
  float* out = (float*)d_out;

  const int NFELEMS = NENV * NNODES * NFD;   // 1,600,000 floats
  float* store = (float*)d_ws;
  float* bufA  = store + NFELEMS;

  const float* cur = nf_in;
  for (int it = 0; it < 3; ++it) {
    gnn_zero_kernel<<<(NFELEMS + 255) / 256, 256, 0, stream>>>(store, NFELEMS);
    gnn_msg_kernel<<<1250, 256, 0, stream>>>(cur, edges, W1, b1, W2, b2, W3, b3, store);
    float* nxt = (it == 1) ? bufA : out;
    gnn_gru_kernel<<<(NENV * NNODES + 255) / 256, 256, 0, stream>>>(
        store, cur, Wih, Whh, bih, bhh, nxt);
    cur = nxt;
  }
}
